// LinearAtten_19988777796308
// MI455X (gfx1250) — matrix operand order repaired and checked by reading, not compiled, not run on hardware
//
#include <hip/hip_runtime.h>
#include <hip/hip_bf16.h>
#include <math.h>

typedef __bf16 bf16;
typedef __attribute__((ext_vector_type(16))) __bf16 v16bf;
typedef __attribute__((ext_vector_type(8)))  __bf16 v8bf;
typedef __attribute__((ext_vector_type(2)))  __bf16 v2bf;
typedef __attribute__((ext_vector_type(8)))  float  v8f;
typedef __attribute__((ext_vector_type(2)))  float  v2f;
typedef __attribute__((ext_vector_type(4)))  unsigned v4u;
typedef __attribute__((ext_vector_type(8)))  int    v8i;
typedef __attribute__((ext_vector_type(4)))  int    v4i;

#define DB   4
#define DQ   4096
#define DC   1024
#define DH   16
#define DHD  64
#define NTOK (DB * DQ)      // 16384
#define KSPLIT 16

// LDS tile geometry for the big GEMMs
#define BK        64
#define LDS_PITCH 72                    // 64 bf16 + 16B pad per row
#define B_SLAB_BYTES (128 * LDS_PITCH * 2)

#if defined(__has_builtin)
#if __has_builtin(__builtin_amdgcn_tensor_load_to_lds) && \
    __has_builtin(__builtin_amdgcn_s_wait_tensorcnt)
#define HAVE_TDM 1
#endif
#endif
#ifndef HAVE_TDM
#define HAVE_TDM 0
#endif

// ---------- helpers ----------

static __device__ inline v16bf cat8(v8bf a, v8bf b) {
  v16bf r;
#pragma unroll
  for (int i = 0; i < 8; ++i) { r[i] = a[i]; r[i + 8] = b[i]; }
  return r;
}

// 16-bit A fragment (16x32): lane<16 holds K{k..k+7, k+16..k+23},
// lane>=16 holds K{k+8..k+15, k+24..k+31}. p = rowBase + kk + laneHi*8.
static __device__ inline v16bf a_frag_from_bf16(const bf16* p) {
  v8bf a = *(const v8bf*)(p);
  v8bf b = *(const v8bf*)(p + 16);
  return cat8(a, b);
}

static __device__ inline v8f wmma_bf16(v16bf a, v16bf b, v8f c) {
  return __builtin_amdgcn_wmma_f32_16x16x32_bf16(false, a, false, b, (short)0, c,
                                                 false, false);
}

static __device__ inline float gelu_exact(float x) {
  return 0.5f * x * (1.0f + erff(x * 0.70710678118654752f));
}

// ---------- weight conversion ----------

__global__ __launch_bounds__(256) void cvt_f32_bf16_kernel(
    const float* __restrict__ src, bf16* __restrict__ dst, int n) {
  int i = blockIdx.x * 256 + threadIdx.x;
  if (i < n) dst[i] = (bf16)src[i];
}

// ---------- generic  out = A(f32, MxK) @ W^T(bf16, NxK) + bias ----------
// LDS-tiled: A staged via regs (f32->bf16 cvt once per block), B staged by the
// Tensor Data Mover into a double-buffered padded LDS tile.
// EPI: 0 = store f32; 1 = GELU then store f32; 2 = add resid then store f32.

template <int EPI>
__global__ __launch_bounds__(256) void gemm_xw_kernel(
    const float* __restrict__ A, const bf16* __restrict__ W,
    const float* __restrict__ bias, const float* __restrict__ resid,
    float* __restrict__ out, int M, int N, int K) {
  __shared__ __align__(16) bf16 lds_a[128 * LDS_PITCH];
  __shared__ __align__(16) bf16 lds_b[2][128 * LDS_PITCH];

  const int tid    = threadIdx.x;
  const int lane   = tid & 31;
  const int wv     = tid >> 5;            // 0..7
  const int laneLo = lane & 15;
  const int laneHi = lane >> 4;
  const int waveM  = wv >> 1;             // 0..3 -> 32-row strip
  const int waveN  = wv & 1;              // 0..1 -> 64-col strip
  const int mBase  = blockIdx.y * 128;
  const int nBase  = blockIdx.x * 128;

  // ---- A slab staging: each thread owns 32 f32 of the 128x64 slab ----
  const int aRow = tid >> 1;
  const int aSeg = (tid & 1) * 32;
  v8f ar[4];
  auto loadA = [&](int kk) {
    const float* p = A + (size_t)(mBase + aRow) * K + kk + aSeg;
#pragma unroll
    for (int j = 0; j < 4; ++j) ar[j] = *(const v8f*)(p + j * 8);
  };
  auto storeA = [&]() {
    bf16* d = &lds_a[aRow * LDS_PITCH + aSeg];
#pragma unroll
    for (int j = 0; j < 4; ++j) {
      v8bf t;
#pragma unroll
      for (int i = 0; i < 8; ++i) t[i] = (__bf16)ar[j][i];
      *(v8bf*)(d + j * 8) = t;
    }
  };

#if HAVE_TDM
  // ---- Tensor DMA descriptor for the B slab (128 rows x 64 bf16, row stride K,
  // LDS rows padded by 16B -> pitch 72 bf16) ----
  const unsigned ldsBBase = (unsigned)(uintptr_t)(void*)&lds_b[0][0];
  v8i g1;
  {
    const unsigned td1 = 1u << 20;  // generous tensor_dim1 (no OOB clipping)
    g1[0] = (int)((1u << 16)        // data_size = 1 -> 2 bytes
                  | (1u << 20)      // pad_enable
                  | (4u << 22)      // pad_interval: 32 DWORDs (128B = one row)
                  | (3u << 25));    // pad_amount: 4 DWORDs (16B)
    g1[1] = (int)(((unsigned)K & 0xFFFFu) << 16);           // tensor_dim0 lo16
    g1[2] = (int)((((unsigned)K >> 16) & 0xFFFFu)           // tensor_dim0 hi16
                  | ((td1 & 0xFFFFu) << 16));               // tensor_dim1 lo16
    g1[3] = (int)(((td1 >> 16) & 0xFFFFu) | (64u << 16));   // dim1 hi | tile_dim0
    g1[4] = (int)128u;                                      // tile_dim1
    g1[5] = (int)(unsigned)K;                               // tensor_dim0_stride lo32
    g1[6] = 0;
    g1[7] = 0;
  }
  v4i g2 = {0, 0, 0, 0};
  v4i g3 = {0, 0, 0, 0};
  auto prefetchB = [&](int kk, int buf) {   // wave0 only
    unsigned long long gaddr =
        (unsigned long long)(uintptr_t)(W + (size_t)nBase * K + kk);
    v4u g0;
    g0[0] = 1u;                                             // count = 1
    g0[1] = ldsBBase + (unsigned)buf * (unsigned)B_SLAB_BYTES;
    g0[2] = (unsigned)(gaddr & 0xFFFFFFFFull);
    g0[3] = (unsigned)((gaddr >> 32) & 0x01FFFFFFull) | (2u << 30);  // type=2
#if __clang_major__ >= 23
    v8i g4 = {0, 0, 0, 0, 0, 0, 0, 0};
    __builtin_amdgcn_tensor_load_to_lds(g0, g1, g2, g3, g4, 0);
#else
    __builtin_amdgcn_tensor_load_to_lds(g0, g1, g2, g3, 0);
#endif
  };
  auto commitB = [&]() { __builtin_amdgcn_s_wait_tensorcnt(0); };  // wave0 only
#else
  // Fallback: cooperative B staging through registers (mirrors the A path).
  v8bf br[4];
  auto prefetchB = [&](int kk, int /*buf*/) {
    const bf16* p = W + (size_t)(nBase + aRow) * K + kk + aSeg;
#pragma unroll
    for (int j = 0; j < 4; ++j) br[j] = *(const v8bf*)(p + j * 8);
  };
  int commitBuf = 0;
  auto commitB = [&]() {
    bf16* d = &lds_b[commitBuf][aRow * LDS_PITCH + aSeg];
#pragma unroll
    for (int j = 0; j < 4; ++j) *(v8bf*)(d + j * 8) = br[j];
  };
#endif

  v8f acc[2][4];
#pragma unroll
  for (int i = 0; i < 2; ++i)
#pragma unroll
    for (int j = 0; j < 4; ++j)
      acc[i][j] = v8f{0.f, 0.f, 0.f, 0.f, 0.f, 0.f, 0.f, 0.f};

  // ---- prologue: slab 0 into LDS ----
  loadA(0);
  storeA();
#if HAVE_TDM
  if (wv == 0) { prefetchB(0, 0); commitB(); }
#else
  prefetchB(0, 0);
  commitBuf = 0;
  commitB();
#endif
  __syncthreads();

  const int nSlab = K / BK;
  for (int i = 0; i < nSlab; ++i) {
    const int buf = i & 1;
    const bool more = (i + 1) < nSlab;
    if (more) {
      loadA((i + 1) * BK);
#if HAVE_TDM
      if (wv == 0) prefetchB((i + 1) * BK, (i + 1) & 1);
#else
      prefetchB((i + 1) * BK, (i + 1) & 1);
#endif
    }

    // ---- compute slab i from LDS (2 k-steps of 32) ----
#pragma unroll
    for (int ks = 0; ks < BK; ks += 32) {
      v16bf aF[2];
#pragma unroll
      for (int im = 0; im < 2; ++im) {
        const bf16* p =
            &lds_a[(waveM * 32 + im * 16 + laneLo) * LDS_PITCH + ks + laneHi * 8];
        aF[im] = a_frag_from_bf16(p);
      }
      v16bf bF[4];
#pragma unroll
      for (int in_ = 0; in_ < 4; ++in_) {
        const bf16* p = &lds_b[buf][(waveN * 64 + in_ * 16 + laneLo) * LDS_PITCH +
                                    ks + laneHi * 8];
        bF[in_] = cat8(*(const v8bf*)(p), *(const v8bf*)(p + 16));
      }
#pragma unroll
      for (int im = 0; im < 2; ++im)
#pragma unroll
        for (int in_ = 0; in_ < 4; ++in_)
          acc[im][in_] = wmma_bf16(aF[im], bF[in_], acc[im][in_]);
    }

    __syncthreads();
    if (more) {
      storeA();
#if HAVE_TDM
      if (wv == 0) commitB();
#else
      commitBuf = (i + 1) & 1;
      commitB();
#endif
      __syncthreads();
    }
  }

  // ---- epilogue ----
  const int m0 = mBase + waveM * 32;
  const int n0 = nBase + waveN * 64;
#pragma unroll
  for (int im = 0; im < 2; ++im) {
#pragma unroll
    for (int in_ = 0; in_ < 4; ++in_) {
      const int col = n0 + in_ * 16 + laneLo;
      const float bv = bias[col];
#pragma unroll
      for (int v = 0; v < 8; ++v) {
        const int row = m0 + im * 16 + laneHi * 8 + v;
        float val = acc[im][in_][v] + bv;
        if (EPI == 1) val = gelu_exact(val);
        if (EPI == 2) val += resid[(size_t)row * N + col];
        out[(size_t)row * N + col] = val;
      }
    }
  }
}

// ---------- split qkv + layernorm(k,v); q->(B,H,Q,hd) bf16, k,v -> (B,H,hd,Q) bf16 ----------

__global__ __launch_bounds__(256) void split_ln_kernel(
    const float* __restrict__ qkv, const float* __restrict__ kn_g,
    const float* __restrict__ kn_b, const float* __restrict__ vn_g,
    const float* __restrict__ vn_b, bf16* __restrict__ qb,
    bf16* __restrict__ kt, bf16* __restrict__ vt) {
  const int lane = threadIdx.x & 31;
  const int wv   = threadIdx.x >> 5;
  const int wid  = blockIdx.x * 8 + wv;  // one wave per (token, head)
  const int t = wid >> 4;
  const int h = wid & 15;
  const int b  = t >> 12;
  const int qq = t & 4095;
  const int d0 = 2 * lane;

  const float* base = qkv + (size_t)t * (3 * DC) + h * DHD + d0;
  v2f qd = *(const v2f*)(base);
  v2f kd = *(const v2f*)(base + DC);
  v2f vd = *(const v2f*)(base + 2 * DC);

  float ks = kd.x + kd.y;
  float vs = vd.x + vd.y;
#pragma unroll
  for (int o = 16; o > 0; o >>= 1) {
    ks += __shfl_xor(ks, o, 32);
    vs += __shfl_xor(vs, o, 32);
  }
  const float km = ks * (1.0f / 64.0f);
  const float vm = vs * (1.0f / 64.0f);
  const float kx0 = kd.x - km, kx1 = kd.y - km;
  const float vx0 = vd.x - vm, vx1 = vd.y - vm;
  float kvr = kx0 * kx0 + kx1 * kx1;
  float vvr = vx0 * vx0 + vx1 * vx1;
#pragma unroll
  for (int o = 16; o > 0; o >>= 1) {
    kvr += __shfl_xor(kvr, o, 32);
    vvr += __shfl_xor(vvr, o, 32);
  }
  const float krs = rsqrtf(kvr * (1.0f / 64.0f) + 1e-5f);
  const float vrs = rsqrtf(vvr * (1.0f / 64.0f) + 1e-5f);

  const float k0 = kx0 * krs * kn_g[d0] + kn_b[d0];
  const float k1 = kx1 * krs * kn_g[d0 + 1] + kn_b[d0 + 1];
  const float v0 = vx0 * vrs * vn_g[d0] + vn_b[d0];
  const float v1 = vx1 * vrs * vn_g[d0 + 1] + vn_b[d0 + 1];

  const int bh = b * DH + h;
  v2bf qp; qp.x = (bf16)qd.x; qp.y = (bf16)qd.y;
  *(v2bf*)(qb + ((size_t)bh * DQ + qq) * DHD + d0) = qp;

  bf16* kp = kt + ((size_t)bh * DHD + d0) * DQ + qq;
  kp[0]  = (bf16)k0;
  kp[DQ] = (bf16)k1;
  bf16* vp = vt + ((size_t)bh * DHD + d0) * DQ + qq;
  vp[0]  = (bf16)v0;
  vp[DQ] = (bf16)v1;
}

// ---------- kv = k^T @ v per (b,h), K-split with f32 atomics ----------

__global__ __launch_bounds__(256) void kv_gemm_kernel(
    const bf16* __restrict__ kt, const bf16* __restrict__ vt,
    float* __restrict__ kv_acc) {
  const int bh = blockIdx.x;
  const int ks = blockIdx.y;
  const int lane   = threadIdx.x & 31;
  const int wv     = threadIdx.x >> 5;
  const int laneLo = lane & 15;
  const int laneHi = lane >> 4;
  const int tm  = (wv >> 1) * 16;
  const int tn0 = (wv & 1) * 32;

  const bf16* ktb = kt + (size_t)bh * DHD * DQ;
  const bf16* vtb = vt + (size_t)bh * DHD * DQ;

  v8f acc[2];
  acc[0] = v8f{0.f, 0.f, 0.f, 0.f, 0.f, 0.f, 0.f, 0.f};
  acc[1] = acc[0];

  const int q0 = ks * (DQ / KSPLIT);
  for (int q = q0; q < q0 + DQ / KSPLIT; q += 32) {
    const bf16* ap = ktb + (size_t)(tm + laneLo) * DQ + q + laneHi * 8;
    v16bf aF = a_frag_from_bf16(ap);
#pragma unroll
    for (int tN = 0; tN < 2; ++tN) {
      const bf16* bp = vtb + (size_t)(tn0 + tN * 16 + laneLo) * DQ + q + laneHi * 8;
      v16bf bF = cat8(*(const v8bf*)(bp), *(const v8bf*)(bp + 16));
      acc[tN] = wmma_bf16(aF, bF, acc[tN]);
    }
  }

#pragma unroll
  for (int tN = 0; tN < 2; ++tN) {
    const int e = tn0 + tN * 16 + laneLo;
#pragma unroll
    for (int v = 0; v < 8; ++v) {
      const int d = tm + laneHi * 8 + v;
      atomicAdd(&kv_acc[(size_t)bh * 4096 + d * 64 + e], acc[tN][v]);
    }
  }
}

__global__ __launch_bounds__(256) void kv_finalize_kernel(
    const float* __restrict__ acc, const float* __restrict__ temp,
    const float* __restrict__ bias, bf16* __restrict__ kvt) {
  const int i = blockIdx.x * 256 + threadIdx.x;
  const int bh = i >> 12;
  const int de = i & 4095;
  const int d = de >> 6;
  const int e = de & 63;
  const int h = bh & 15;
  const float val = acc[i] * (1.0f / (float)DQ) * temp[h] + bias[h * 4096 + de];
  kvt[(size_t)bh * 4096 + e * 64 + d] = (bf16)val;
}

// ---------- att = q @ kv, fused "+x" residual -> (B,Q,C) f32 ----------

__global__ __launch_bounds__(256) void att_gemm_kernel(
    const bf16* __restrict__ qb, const bf16* __restrict__ kvt,
    const float* __restrict__ x, float* __restrict__ att) {
  const int bh = blockIdx.y;
  const int q0 = blockIdx.x * 128;
  const int b = bh >> 4, h = bh & 15;
  const int lane   = threadIdx.x & 31;
  const int wv     = threadIdx.x >> 5;
  const int laneLo = lane & 15;
  const int laneHi = lane >> 4;

  v8f acc[4];
#pragma unroll
  for (int t = 0; t < 4; ++t) acc[t] = v8f{0.f, 0.f, 0.f, 0.f, 0.f, 0.f, 0.f, 0.f};

#pragma unroll
  for (int kk = 0; kk < DHD; kk += 32) {
    const bf16* ap =
        qb + ((size_t)bh * DQ + q0 + wv * 16 + laneLo) * DHD + kk + laneHi * 8;
    v16bf aF = a_frag_from_bf16(ap);
#pragma unroll
    for (int t = 0; t < 4; ++t) {
      const bf16* bp =
          kvt + (size_t)bh * 4096 + (t * 16 + laneLo) * 64 + kk + laneHi * 8;
      v16bf bF = cat8(*(const v8bf*)(bp), *(const v8bf*)(bp + 16));
      acc[t] = wmma_bf16(aF, bF, acc[t]);
    }
  }

#pragma unroll
  for (int t = 0; t < 4; ++t) {
    const int col = t * 16 + laneLo;
#pragma unroll
    for (int v = 0; v < 8; ++v) {
      const int row = q0 + wv * 16 + laneHi * 8 + v;
      const size_t oi = ((size_t)(b * DQ + row)) * DC + h * DHD + col;
      att[oi] = acc[t][v] + x[oi];
    }
  }
}

// ---------- launch ----------

extern "C" void kernel_launch(void* const* d_in, const int* in_sizes, int n_in,
                              void* d_out, int out_size, void* d_ws, size_t ws_size,
                              hipStream_t stream) {
  (void)in_sizes; (void)n_in; (void)out_size; (void)ws_size;
  const float* x     = (const float*)d_in[0];
  const float* qkv_w = (const float*)d_in[1];
  const float* qkv_b = (const float*)d_in[2];
  const float* kn_g  = (const float*)d_in[3];
  const float* kn_b  = (const float*)d_in[4];
  const float* vn_g  = (const float*)d_in[5];
  const float* vn_b  = (const float*)d_in[6];
  const float* temp  = (const float*)d_in[7];
  const float* abias = (const float*)d_in[8];
  const float* p1_w  = (const float*)d_in[9];
  const float* p1_b  = (const float*)d_in[10];
  const float* p2_w  = (const float*)d_in[11];
  const float* p2_b  = (const float*)d_in[12];
  float* out = (float*)d_out;

  char* ws = (char*)d_ws;
  size_t off = 0;
  float* qkv_f32 = (float*)(ws + off);
  float* att_res = (float*)(ws + off);                         // alias (after LN)
  float* hbuf    = (float*)(ws + off + (size_t)NTOK * DC * 4); // alias (after LN)
  off += (size_t)NTOK * 3 * DC * 4;

  bf16* qkvw = (bf16*)(ws + off); off += (size_t)3 * DC * DC * 2;
  bf16* p1w  = (bf16*)(ws + off); off += (size_t)DC * DC * 2;
  bf16* p2w  = (bf16*)(ws + off); off += (size_t)DC * DC * 2;
  bf16* qb   = (bf16*)(ws + off); off += (size_t)NTOK * DC * 2;
  bf16* kt   = (bf16*)(ws + off); off += (size_t)NTOK * DC * 2;
  bf16* vt   = (bf16*)(ws + off); off += (size_t)NTOK * DC * 2;
  float* kv_acc = (float*)(ws + off); off += (size_t)64 * 64 * 64 * 4;
  bf16* kvt  = (bf16*)(ws + off); off += (size_t)64 * 64 * 64 * 2;

  const dim3 blk(256);

  cvt_f32_bf16_kernel<<<(3 * DC * DC + 255) / 256, blk, 0, stream>>>(qkv_w, qkvw, 3 * DC * DC);
  cvt_f32_bf16_kernel<<<(DC * DC + 255) / 256, blk, 0, stream>>>(p1_w, p1w, DC * DC);
  cvt_f32_bf16_kernel<<<(DC * DC + 255) / 256, blk, 0, stream>>>(p2_w, p2w, DC * DC);

  gemm_xw_kernel<0><<<dim3(3 * DC / 128, NTOK / 128), blk, 0, stream>>>(
      x, qkvw, qkv_b, nullptr, qkv_f32, NTOK, 3 * DC, DC);

  split_ln_kernel<<<NTOK * DH / 8, blk, 0, stream>>>(qkv_f32, kn_g, kn_b, vn_g,
                                                     vn_b, qb, kt, vt);

  hipMemsetAsync(kv_acc, 0, (size_t)64 * 64 * 64 * 4, stream);
  kv_gemm_kernel<<<dim3(64, KSPLIT), blk, 0, stream>>>(kt, vt, kv_acc);
  kv_finalize_kernel<<<(64 * 64 * 64) / 256, blk, 0, stream>>>(kv_acc, temp, abias, kvt);

  att_gemm_kernel<<<dim3(DQ / 128, 64), blk, 0, stream>>>(qb, kvt, x, att_res);

  gemm_xw_kernel<1><<<dim3(DC / 128, NTOK / 128), blk, 0, stream>>>(
      att_res, p1w, p1_b, nullptr, hbuf, NTOK, DC, DC);

  gemm_xw_kernel<2><<<dim3(DC / 128, NTOK / 128), blk, 0, stream>>>(
      hbuf, p2w, p2_b, x, out, NTOK, DC, DC);
}
